// ContraNorm_41721312313814
// MI455X (gfx1250) — compile-verified, matrix operand order checked by reading, not run on hardware
//
#include <hip/hip_runtime.h>
#include <hip/hip_bf16.h>

// ---------------------------------------------------------------------------
// ContraNorm fused flash-style kernel for gfx1250 (MI455X), wave32 + WMMA +
// double-buffered Tensor Data Mover staging.
//   sim = (x/||x||) @ (x/||x||)^T ; sim[edges] = -inf
//   attn = softmax(sim, dim=1)    ; out = 1.1*x - 0.1*attn@x
// cos-sim <= 1  =>  softmax with fixed offset p = exp(s-1): no online max.
// ---------------------------------------------------------------------------

#define CN_N 8192
#define CN_D 256
#define CN_E 262144
#define CN_MASKW (CN_N / 32)   // 256 u32 words per row
#define CN_SCALE 0.1f

typedef __attribute__((ext_vector_type(16))) _Float16 v16h;
typedef __attribute__((ext_vector_type(8)))  _Float16 v8h;
typedef __attribute__((ext_vector_type(8)))  float    v8f;
typedef __attribute__((ext_vector_type(4)))  unsigned int u32x4;
typedef __attribute__((ext_vector_type(8)))  int      i32x8;
typedef __attribute__((ext_vector_type(4)))  int      i32x4;

__device__ __forceinline__ v16h cn_cat(v8h lo, v8h hi) {
    return __builtin_shufflevector(lo, hi, 0,1,2,3,4,5,6,7,8,9,10,11,12,13,14,15);
}
__device__ __forceinline__ v16h cn_lds_a(const _Float16* p) {
    return cn_cat(*(const v8h*)p, *(const v8h*)(p + 16));
}

// ------------------------------------------------------------------ TDM load
// Issue a 2D tensor DMA (global -> LDS) per CDNA5 D# layout (ISA ch. 8.3-8.6).
// data_size = 2 bytes. LDS padding: pad (pad_amt+1) DWORDs after each
// 2^(pad_iv+1) DWORDs stored (encodes our bank-conflict padding for free).
__device__ __forceinline__ void cn_tdm_load_2d(
    const void* gaddr, unsigned lds_off,
    unsigned tensor_d0, unsigned tensor_d1,
    unsigned tile_d0,   unsigned tile_d1,
    unsigned long long stride0,
    unsigned pad_iv, unsigned pad_amt, bool pad_en)
{
    const unsigned long long ga = (unsigned long long)(uintptr_t)gaddr;
    u32x4 g0;
    g0[0] = 1u;                                             // count=1 (valid)
    g0[1] = lds_off;                                        // lds_addr
    g0[2] = (unsigned)(ga & 0xFFFFFFFFu);                   // global_addr lo
    g0[3] = (unsigned)((ga >> 32) & 0x01FFFFFFu) | (2u << 30); // hi | type=2
    i32x8 g1;
    unsigned w0 = (1u << 16);                               // data_size=2B
    if (pad_en) w0 |= (1u << 20) | (pad_iv << 22) | (pad_amt << 25);
    g1[0] = (int)w0;                                        // wg_mask=0 etc.
    g1[1] = (int)((tensor_d0 & 0xFFFFu) << 16);             // tdim0[15:0]
    g1[2] = (int)((tensor_d0 >> 16) | ((tensor_d1 & 0xFFFFu) << 16));
    g1[3] = (int)((tensor_d1 >> 16) | (tile_d0 << 16));     // tile_dim0
    g1[4] = (int)(tile_d1 & 0xFFFFu);                       // tile_dim1, d2=0
    g1[5] = (int)(unsigned)(stride0 & 0xFFFFFFFFu);         // stride0 lo
    g1[6] = (int)(unsigned)((stride0 >> 32) & 0xFFFFu);     // stride0 hi
    g1[7] = 0;
    const i32x4 z4 = {0, 0, 0, 0};                          // groups 2/3: 2D
    const i32x8 z8 = {0, 0, 0, 0, 0, 0, 0, 0};
    __builtin_amdgcn_tensor_load_to_lds(g0, g1, z4, z4, z8, 0);
}

// ---------------------------------------------------------------- prep: zero
__global__ void cn_zero_kernel(unsigned* __restrict__ p, int n) {
    int i = blockIdx.x * blockDim.x + threadIdx.x;
    int stride = gridDim.x * blockDim.x;
    for (; i < n; i += stride) p[i] = 0u;
}

// ------------------------------------------------- prep: norms + f16 layouts
__global__ __launch_bounds__(256)
void cn_norm_kernel(const float* __restrict__ x,
                    _Float16* __restrict__ normx,
                    _Float16* __restrict__ xT) {
    const int row = blockIdx.x;
    const int d   = threadIdx.x;
    const float v = x[row * CN_D + d];
    float s = v * v;
    #pragma unroll
    for (int m = 16; m >= 1; m >>= 1) s += __shfl_xor(s, m, 32);
    __shared__ float red[8];
    if ((threadIdx.x & 31) == 0) red[threadIdx.x >> 5] = s;
    __syncthreads();
    float tot = 0.f;
    #pragma unroll
    for (int w = 0; w < 8; ++w) tot += red[w];
    const float inv = 1.0f / fmaxf(sqrtf(tot), 1e-12f);
    normx[row * CN_D + d] = (_Float16)(v * inv);
    xT[d * CN_N + row]    = (_Float16)v;
}

// ------------------------------------------------------ prep: edge bit mask
__global__ __launch_bounds__(256)
void cn_scatter_kernel(const long long* __restrict__ ei,
                       unsigned* __restrict__ mask) {
    const int t = blockIdx.x * blockDim.x + threadIdx.x;
    if (t < CN_E) {
        const int r = (int)ei[t];
        const int c = (int)ei[CN_E + t];
        atomicOr(&mask[r * CN_MASKW + (c >> 5)], 1u << (c & 31));
    }
}

// --------------------------------------------------------------- fused core
// 4 waves / block; wave w owns 16 output rows. TDM double-buffers both tiles
// (DMA for jb+32 overlaps compute on jb); per j-iteration per wave:
// 16 WMMA for S^T + 16 WMMA for PV.
__global__ __launch_bounds__(128)
void cn_fused_kernel(const float* __restrict__ x,
                     const unsigned* __restrict__ mask,
                     const _Float16* __restrict__ normx,
                     const _Float16* __restrict__ xT,
                     float* __restrict__ out) {
    __shared__ _Float16 lds_nx[2][32][CN_D + 8];   // normx rows j..j+31
    __shared__ _Float16 lds_xt[2][CN_D][32 + 8];   // xT[d][j..j+31]

    const int tid    = threadIdx.x;
    const int wave   = tid >> 5;
    const int lane   = tid & 31;
    const int laneLo = lane & 15;
    const int laneHi = lane >> 4;
    const int ibase  = (blockIdx.x * 4 + wave) * 16;
    const int irow   = ibase + laneLo;

    const unsigned nx_lds[2] = {(unsigned)(uintptr_t)&lds_nx[0][0][0],
                                (unsigned)(uintptr_t)&lds_nx[1][0][0]};
    const unsigned xt_lds[2] = {(unsigned)(uintptr_t)&lds_xt[0][0][0],
                                (unsigned)(uintptr_t)&lds_xt[1][0][0]};

    auto issue_tiles = [&](int jb, int buf) {
        // normx rows jb..jb+31: flat 8192 halfs, pad 4 DW per 128 DW
        cn_tdm_load_2d(normx + (size_t)jb * CN_D, nx_lds[buf],
                       8192u, 1u, 8192u, 1u, 8192ull,
                       /*pad_iv=*/6u, /*pad_amt=*/3u, true);
        // xT[0..255][jb..jb+31]: 32 x 256 tile, stride N, pad 4 DW / 16 DW
        cn_tdm_load_2d(xT + jb, xt_lds[buf],
                       (unsigned)CN_N, (unsigned)CN_D, 32u, (unsigned)CN_D,
                       (unsigned long long)CN_N,
                       /*pad_iv=*/3u, /*pad_amt=*/3u, true);
    };

    // Resident B operand for S^T: normx rows i, K=256 as 8 chunks of 32.
    v16h Bi[8];
    #pragma unroll
    for (int kc = 0; kc < 8; ++kc) {
        const _Float16* p = normx + irow * CN_D + kc * 32 + laneHi * 8;
        Bi[kc] = cn_cat(*(const v8h*)p, *(const v8h*)(p + 16));
    }

    v8f acc[16];                       // x_neg^T accumulators (16 d-tiles)
    #pragma unroll
    for (int dt = 0; dt < 16; ++dt) acc[dt] = (v8f)0.0f;
    float lsum = 0.0f;

    if (wave == 0) issue_tiles(0, 0);  // prologue DMA into buffer 0

    int cur = 0;
    for (int jb = 0; jb < CN_N; jb += 32, cur ^= 1) {
        // TENSORcnt is in-order per wave: the only outstanding tensor loads
        // here are the two for buffer `cur` -> wait(0) publishes exactly them.
        if (wave == 0) __builtin_amdgcn_s_wait_tensorcnt(0);
        __syncthreads();   // cur ready for all; all done reading buf cur^1
        if (wave == 0 && jb + 32 < CN_N) issue_tiles(jb + 32, cur ^ 1);

        const unsigned mw = mask[irow * CN_MASKW + (jb >> 5)];
        __builtin_prefetch(mask + irow * CN_MASKW + ((jb >> 5) + 1), 0, 3);

        // ---- S^T tiles (j x i), contraction over D, software-pipelined
        v8f s0 = (v8f)0.0f, s1 = (v8f)0.0f;
        v16h a0 = cn_lds_a(&lds_nx[cur][laneLo][laneHi * 8]);
        v16h a1 = cn_lds_a(&lds_nx[cur][16 + laneLo][laneHi * 8]);
        #pragma unroll
        for (int kc = 0; kc < 8; ++kc) {
            v16h n0 = a0, n1 = a1;
            if (kc < 7) {
                n0 = cn_lds_a(&lds_nx[cur][laneLo][(kc + 1) * 32 + laneHi * 8]);
                n1 = cn_lds_a(&lds_nx[cur][16 + laneLo][(kc + 1) * 32 + laneHi * 8]);
            }
            s0 = __builtin_amdgcn_wmma_f32_16x16x32_f16(
                     false, a0, false, Bi[kc], (short)0, s0, false, false);
            s1 = __builtin_amdgcn_wmma_f32_16x16x32_f16(
                     false, a1, false, Bi[kc], (short)0, s1, false, false);
            a0 = n0; a1 = n1;
        }

        // ---- masked exp; S^T C-layout packs directly into PV B-operand
        v16h Bp;
        #pragma unroll
        for (int r = 0; r < 8; ++r) {
            const int b0 = r + laneHi * 8;
            const float p0 = ((mw >> b0) & 1u) ? 0.0f : __expf(s0[r] - 1.0f);
            const float p1 = ((mw >> (b0 + 16)) & 1u) ? 0.0f : __expf(s1[r] - 1.0f);
            lsum += p0 + p1;
            Bp[r]     = (_Float16)p0;
            Bp[r + 8] = (_Float16)p1;
        }

        // ---- PV: acc[dt] += xT_tile (d x 32j) * P^T (32j x i), pipelined
        v16h ax = cn_lds_a(&lds_xt[cur][laneLo][laneHi * 8]);
        #pragma unroll
        for (int dt = 0; dt < 16; ++dt) {
            v16h nx = ax;
            if (dt < 15)
                nx = cn_lds_a(&lds_xt[cur][(dt + 1) * 16 + laneLo][laneHi * 8]);
            acc[dt] = __builtin_amdgcn_wmma_f32_16x16x32_f16(
                          false, ax, false, Bp, (short)0, acc[dt], false, false);
            ax = nx;
        }
    }

    // ---- epilogue: finish denominator, write out as b128 stores
    const float ltot = lsum + __shfl_xor(lsum, 16, 32);
    const float coef = CN_SCALE / ltot;
    #pragma unroll
    for (int dt = 0; dt < 16; ++dt) {
        const int d0 = dt * 16 + laneHi * 8;          // C-layout: M = d
        const float4* xs = (const float4*)(x + irow * CN_D + d0);
        const float4 xa = xs[0], xb = xs[1];
        float4 oa, ob;
        oa.x = (1.0f + CN_SCALE) * xa.x - coef * acc[dt][0];
        oa.y = (1.0f + CN_SCALE) * xa.y - coef * acc[dt][1];
        oa.z = (1.0f + CN_SCALE) * xa.z - coef * acc[dt][2];
        oa.w = (1.0f + CN_SCALE) * xa.w - coef * acc[dt][3];
        ob.x = (1.0f + CN_SCALE) * xb.x - coef * acc[dt][4];
        ob.y = (1.0f + CN_SCALE) * xb.y - coef * acc[dt][5];
        ob.z = (1.0f + CN_SCALE) * xb.z - coef * acc[dt][6];
        ob.w = (1.0f + CN_SCALE) * xb.w - coef * acc[dt][7];
        float4* os = (float4*)(out + irow * CN_D + d0);
        os[0] = oa;
        os[1] = ob;
    }
}

// ---------------------------------------------------------------------------
extern "C" void kernel_launch(void* const* d_in, const int* in_sizes, int n_in,
                              void* d_out, int out_size, void* d_ws, size_t ws_size,
                              hipStream_t stream) {
    const float*     x  = (const float*)d_in[0];
    const long long* ei = (const long long*)d_in[1];
    float* out = (float*)d_out;

    unsigned char* ws = (unsigned char*)d_ws;
    unsigned* mask  = (unsigned*)ws;                               // 8 MB
    _Float16* normx = (_Float16*)(ws + (size_t)8  * 1024 * 1024);  // 4 MB
    _Float16* xT    = (_Float16*)(ws + (size_t)12 * 1024 * 1024);  // 4 MB

    cn_zero_kernel<<<2048, 256, 0, stream>>>(mask, CN_N * CN_MASKW);
    cn_norm_kernel<<<CN_N, 256, 0, stream>>>(x, normx, xT);
    cn_scatter_kernel<<<CN_E / 256, 256, 0, stream>>>(ei, mask);
    cn_fused_kernel<<<CN_N / 64, 128, 0, stream>>>(x, mask, normx, xT, out);
}
